// EGT_27616639713835
// MI455X (gfx1250) — compile-verified, matrix operand order checked by reading, not run on hardware
//
#include <hip/hip_runtime.h>
#include <math.h>

// EGT fused attention for MI455X (gfx1250, wave32).
// B=4, N=1024, H=8, D=32. Bandwidth-bound (~540MB min traffic -> ~23us @ 23.3TB/s).
// One WG = one (b, h, 16-row l tile); full 16x1024 score row + sig(G) resident in LDS.

#define NB 4
#define NN 1024
#define NH 8
#define ND 32
#define C3 (3 * ND * NH)   // 768 floats per (b,n) row of QKV
#define CLIP_LO -5.0f
#define CLIP_HI  5.0f

typedef __attribute__((ext_vector_type(16))) __bf16 v16bf;
typedef __attribute__((ext_vector_type(8)))  float  v8f;

__device__ __forceinline__ size_t eg_idx(int b, int l, int m, int h) {
    return (((size_t)b * NN + l) * NN + m) * NH + h;
}

__global__ __launch_bounds__(128)
void egt_fused(const float* __restrict__ QKV,
               const float* __restrict__ E,
               const float* __restrict__ G,
               float* __restrict__ Vout,
               float* __restrict__ Hout,
               float* __restrict__ Aout) {
    __shared__ float s_scores[16][NN];   // 64 KB: H_hat row block
    __shared__ float s_sigg[16][NN];     // 64 KB: sigmoid(G) row block
    __shared__ float s_q[16][ND];        // Q tile
    __shared__ float s_red[16][8];
    __shared__ float s_red2[16][8];
    __shared__ float s_mx[16], s_iz[16], s_ds[16];
    __shared__ float s_vacc[4][16][ND];  // per-wave V_att partials

    const int tid  = threadIdx.x;
    const int wave = tid >> 5;
    const int lane = tid & 31;
    const int Mrow = lane & 15;          // A-frag row / B-frag col / C-frag col
    const int klo  = (lane >> 4) * 8;    // K-offset half of lane group (ISA 7.12.2)

    const int bid = blockIdx.x;          // h fastest -> 8 head-WGs share E/G lines in L2
    const int h   = bid & 7;
    const int lt  = (bid >> 3) & 63;
    const int b   = bid >> 9;
    const int l0  = lt * 16;

    const size_t qkv_b = (size_t)b * NN * C3;

    // ---- load Q tile (16 x 32) into LDS ----
    for (int e = tid; e < 16 * ND; e += 128) {
        int l = e >> 5, d = e & 31;
        s_q[l][d] = QKV[qkv_b + (size_t)(l0 + l) * C3 + d * NH + h];
    }
    __syncthreads();

    // ---- build per-wave Q A-fragment (16x32 bf16), reused for all m-chunks ----
    v16bf qa;
#pragma unroll
    for (int t = 0; t < 16; ++t) {
        int K = klo + (t < 8 ? t : t + 8);
        qa[t] = (__bf16)s_q[Mrow][K];
    }

    // ---- phase 1: scores = clip(Q.K^T) + E ; write H_hat ; stash sig(G) ----
    for (int mc = wave; mc < NN / 16; mc += 4) {
        const int m0 = mc * 16;
        v16bf kb;                            // B operand: K^T chunk (32d x 16m)
#pragma unroll
        for (int t = 0; t < 16; ++t) {
            int K = klo + (t < 8 ? t : t + 8);
            kb[t] = (__bf16)QKV[qkv_b + (size_t)(m0 + Mrow) * C3 + (ND * NH) + K * NH + h];
        }
        v8f c = {};
        c = __builtin_amdgcn_wmma_f32_16x16x32_bf16(false, qa, false, kb,
                                                    (short)0, c, false, false);
#pragma unroll
        for (int r = 0; r < 8; ++r) {
            int M = r + klo;                 // C/D layout: M = r + (lane>>4)*8
            int m = m0 + Mrow;               // N = lane & 15
            float s = fminf(CLIP_HI, fmaxf(CLIP_LO, c[r]));
            size_t ix = eg_idx(b, l0 + M, m, h);
            float hv = s + E[ix];
            __builtin_nontemporal_store(hv, &Hout[ix]);   // streamed, never re-read
            s_scores[M][m] = hv;
            float sg = 1.0f / (1.0f + __expf(-G[ix]));
            s_sigg[M][m] = sg;
        }
    }
    __syncthreads();

    // ---- phase 2: per-row softmax stats + degrees (from LDS, no HBM) ----
    {
        int r = tid >> 3, seg = tid & 7;
        const float* row = &s_scores[r][seg * 128];
        float mx = -1e30f;
        for (int c2 = 0; c2 < 128; ++c2) mx = fmaxf(mx, row[c2]);
        s_red[r][seg] = mx;
    }
    __syncthreads();
    if (tid < 16) {
        float mx = s_red[tid][0];
        for (int s2 = 1; s2 < 8; ++s2) mx = fmaxf(mx, s_red[tid][s2]);
        s_mx[tid] = mx;
    }
    __syncthreads();
    {
        int r = tid >> 3, seg = tid & 7;
        float mx = s_mx[r], z = 0.f, dg = 0.f;
        const float* row  = &s_scores[r][seg * 128];
        const float* grow = &s_sigg[r][seg * 128];
        for (int c2 = 0; c2 < 128; ++c2) {
            z  += __expf(row[c2] - mx);
            dg += grow[c2];
        }
        s_red[r][seg] = z; s_red2[r][seg] = dg;
    }
    __syncthreads();
    if (tid < 16) {
        float z = 0.f, dg = 0.f;
        for (int s2 = 0; s2 < 8; ++s2) { z += s_red[tid][s2]; dg += s_red2[tid][s2]; }
        s_iz[tid] = 1.0f / z;
        s_ds[tid] = log1pf(dg);             // degree scaler
    }
    __syncthreads();

    // ---- phase 3: A_tild = softmax * sig(G) -> global; V_att = A_tild . V via WMMA ----
    v8f acc0 = {}, acc1 = {};
    const float mxM = s_mx[Mrow];
    const float izM = s_iz[Mrow];
    for (int mc = wave; mc < NN / 32; mc += 4) {
        const int m0 = mc * 32;
        v16bf a, b0, b1;
#pragma unroll
        for (int t = 0; t < 16; ++t) {
            int K = klo + (t < 8 ? t : t + 8);
            int m = m0 + K;
            float val = __expf(s_scores[Mrow][m] - mxM) * izM * s_sigg[Mrow][m];
            a[t] = (__bf16)val;
            // A-frag covers each (Mrow, m) exactly once -> fuse A_tild store here
            __builtin_nontemporal_store(val, &Aout[eg_idx(b, l0 + Mrow, m, h)]);
            size_t vb = qkv_b + (size_t)m * C3 + 2 * (ND * NH) + h;
            b0[t] = (__bf16)QKV[vb + (size_t)Mrow * NH];          // V cols d = 0..15
            b1[t] = (__bf16)QKV[vb + (size_t)(16 + Mrow) * NH];   // V cols d = 16..31
        }
        acc0 = __builtin_amdgcn_wmma_f32_16x16x32_bf16(false, a, false, b0,
                                                       (short)0, acc0, false, false);
        acc1 = __builtin_amdgcn_wmma_f32_16x16x32_bf16(false, a, false, b1,
                                                       (short)0, acc1, false, false);
    }
#pragma unroll
    for (int r = 0; r < 8; ++r) {
        int M = r + klo;
        s_vacc[wave][M][Mrow]      = acc0[r];
        s_vacc[wave][M][16 + Mrow] = acc1[r];
    }
    __syncthreads();

    // ---- reduce 4 wave-partials, apply log1p(degrees), write V_att ----
    for (int e = tid; e < 16 * ND; e += 128) {
        int l = e >> 5, d = e & 31;
        float v = s_vacc[0][l][d] + s_vacc[1][l][d] + s_vacc[2][l][d] + s_vacc[3][l][d];
        v *= s_ds[l];
        Vout[((size_t)(b * NN + l0 + l)) * (ND * NH) + d * NH + h] = v;
    }
}

extern "C" void kernel_launch(void* const* d_in, const int* in_sizes, int n_in,
                              void* d_out, int out_size, void* d_ws, size_t ws_size,
                              hipStream_t stream) {
    const float* QKV = (const float*)d_in[0];
    const float* E   = (const float*)d_in[1];
    const float* G   = (const float*)d_in[2];
    float* out  = (float*)d_out;
    float* Vout = out;                                     // (4,1024,256)
    float* Hout = Vout + (size_t)NB * NN * ND * NH;        // (4,1024,1024,8)
    float* Aout = Hout + (size_t)NB * NN * NN * NH;        // (4,1024,1024,8)

    dim3 grid(NB * (NN / 16) * NH);   // 2048 WGs, h fastest for L2 line sharing
    egt_fused<<<grid, 128, 0, stream>>>(QKV, E, G, Vout, Hout, Aout);
}
